// TransformEncoder_4801773437669
// MI455X (gfx1250) — compile-verified, matrix operand order checked by reading, not run on hardware
//
#include <hip/hip_runtime.h>
#include <math.h>

// ---------------------------------------------------------------------------
// Problem constants (match reference)
// ---------------------------------------------------------------------------
#define N_NODES 100000
#define N_EDGES 1600000
#define D_IN    256
#define D_H1    128
#define D_H2    16

typedef _Float16 half_t;
typedef __attribute__((ext_vector_type(16))) _Float16 v16h;
typedef __attribute__((ext_vector_type(8)))  _Float16 v8h;
typedef __attribute__((ext_vector_type(8)))  float    v8f;
typedef int v4i __attribute__((vector_size(16)));

#if defined(__has_builtin)
#if __has_builtin(__builtin_amdgcn_global_load_async_to_lds_b128)
#define HAVE_ASYNC_TO_LDS 1
#endif
#if __has_builtin(__builtin_amdgcn_sched_barrier)
#define SCHED_FENCE() __builtin_amdgcn_sched_barrier(0)
#endif
#endif
#ifndef SCHED_FENCE
#define SCHED_FENCE()
#endif

// ---------------------------------------------------------------------------
// Helpers
// ---------------------------------------------------------------------------
__device__ __forceinline__ void atomicMaxF(float* addr, float value) {
    unsigned int* ua = (unsigned int*)addr;
    unsigned int old = *ua;
    while (__uint_as_float(old) < value) {
        unsigned int assumed = old;
        old = atomicCAS(ua, assumed, __float_as_uint(value));
        if (old == assumed) break;
    }
}

__device__ __forceinline__ void wait_asynccnt0() {
#if defined(__has_builtin)
#if __has_builtin(__builtin_amdgcn_s_wait_asynccnt)
    __builtin_amdgcn_s_wait_asynccnt(0);
    return;
#endif
#endif
    asm volatile("s_wait_asynccnt 0" ::: "memory");
}

// ---------------------------------------------------------------------------
// f32 -> f16 conversion (plain)
// ---------------------------------------------------------------------------
__global__ void k_f32_to_f16(const float* __restrict__ src,
                             half_t* __restrict__ dst, int n) {
    int i = blockIdx.x * blockDim.x + threadIdx.x;
    if (i < n) dst[i] = (half_t)src[i];
}

// ---------------------------------------------------------------------------
// Pack 4 weight matrices [K x Dh] (row major) into one transposed f16 matrix
// Wt[(4*Dh) x K], plus concatenated bias[4*Dh].
// ---------------------------------------------------------------------------
__global__ void k_pack_w(const float* __restrict__ W0, const float* __restrict__ W1,
                         const float* __restrict__ W2, const float* __restrict__ W3,
                         const float* __restrict__ B0, const float* __restrict__ B1,
                         const float* __restrict__ B2, const float* __restrict__ B3,
                         half_t* __restrict__ Wt, float* __restrict__ bias,
                         int K, int Dh) {
    int total = 4 * Dh * K;
    int tid = blockIdx.x * blockDim.x + threadIdx.x;
    if (tid < total) {
        int n   = tid / K;      // output row (transposed): column of original W
        int k   = tid % K;
        int sel = n / Dh;
        int col = n % Dh;
        const float* W = (sel == 0) ? W0 : (sel == 1) ? W1 : (sel == 2) ? W2 : W3;
        Wt[(size_t)n * K + k] = (half_t)W[(size_t)k * Dh + col];
    }
    if (tid < 4 * Dh) {
        int sel = tid / Dh;
        int col = tid % Dh;
        const float* B = (sel == 0) ? B0 : (sel == 1) ? B1 : (sel == 2) ? B2 : B3;
        bias[tid] = B[col];
    }
}

// ---------------------------------------------------------------------------
// WMMA GEMM with LDS-staged weights:
//   C[M x N] = A[M x K](f16) * Bt[N x K]^T (f16) + bias[N]
// Block = 8 waves = 128-row M strip x 64-col N strip.
//   - The 64 x K weight slice (shared by all 8 waves) is staged into LDS via
//     the gfx1250 async global->LDS path (ASYNCcnt) when available.
//   - Each wave owns one 16-row M tile and computes 4 N-tiles per k-step,
//     reusing a single A fragment across 4 v_wmma_f32_16x16x32_f16 ops.
//   - A sched_barrier(0) pins all 8 ds_load_b128 ahead of the WMMA group so
//     the 4 matrix ops issue back-to-back behind a single s_wait_dscnt.
// Per-lane fragment layout per CDNA5 ISA 7.12.2 (16-bit A, 16x32):
//   lane<16  : row = lane,    K = {0..7, 16..23}
//   lane>=16 : row = lane-16, K = {8..15, 24..31}
// ---------------------------------------------------------------------------
#define BK_MAX 256
__global__ void k_wmma_gemm(const half_t* __restrict__ A,
                            const half_t* __restrict__ Bt,
                            const float* __restrict__ bias,
                            float* __restrict__ C,
                            int M, int K, int N) {
    __shared__ half_t lds_b[64 * BK_MAX];   // 32 KB max (K <= 256)

    const int n_strips = N >> 6;            // 64-column strips
    const int ms = blockIdx.x / n_strips;   // M-strip index (128 rows)
    const int ns = blockIdx.x - ms * n_strips;
    const int n_base = ns << 6;

    // ---- stage the 64 x K weight slice into LDS (rows contiguous in Bt) ----
    const half_t* bsrc = Bt + (size_t)n_base * K;
    const int chunks = (64 * K) >> 3;       // 16-byte chunks; multiple of 256
#ifdef HAVE_ASYNC_TO_LDS
    {
        typedef __attribute__((address_space(1))) v4i gv4i;
        typedef __attribute__((address_space(3))) v4i lv4i;
        for (int c = threadIdx.x; c < chunks; c += blockDim.x) {
            __builtin_amdgcn_global_load_async_to_lds_b128(
                (gv4i*)(const_cast<half_t*>(bsrc) + (size_t)c * 8),
                (lv4i*)(&lds_b[(size_t)c * 8]), 0, 0);
        }
        wait_asynccnt0();
    }
#else
    for (int c = threadIdx.x; c < chunks; c += blockDim.x)
        *(v8h*)(&lds_b[(size_t)c * 8]) = *(const v8h*)(bsrc + (size_t)c * 8);
#endif
    __syncthreads();

    const int wave = threadIdx.x >> 5;
    const int lane = threadIdx.x & 31;
    const int l15  = lane & 15;
    const int hi   = (lane >> 4) << 3;      // 0 or 8: K sub-offset per half-wave

    // A row for this lane (clamped so remainder strips never read OOB;
    // stores are guarded, EXEC stays all-ones through the WMMAs).
    int mrow = (ms << 7) + (wave << 4) + l15;
    if (mrow >= M) mrow = M - 1;
    const half_t* arow = A + (size_t)mrow * K;
    const half_t* brow0 = lds_b + (size_t)(0  + l15) * K;
    const half_t* brow1 = lds_b + (size_t)(16 + l15) * K;
    const half_t* brow2 = lds_b + (size_t)(32 + l15) * K;
    const half_t* brow3 = lds_b + (size_t)(48 + l15) * K;

    v8f acc0 = {}, acc1 = {}, acc2 = {}, acc3 = {};
#pragma clang loop unroll_count(2)
    for (int k0 = 0; k0 < K; k0 += 32) {
        v8h a0 = *(const v8h*)(arow + k0 + hi);
        v8h a1 = *(const v8h*)(arow + k0 + 16 + hi);
        v16h av = __builtin_shufflevector(a0, a1,
                    0,1,2,3,4,5,6,7,8,9,10,11,12,13,14,15);

        // all 8 ds_load_b128 issued here, pinned by the sched barrier below
        v8h b00 = *(const v8h*)(brow0 + k0 + hi);
        v8h b01 = *(const v8h*)(brow0 + k0 + 16 + hi);
        v8h b10 = *(const v8h*)(brow1 + k0 + hi);
        v8h b11 = *(const v8h*)(brow1 + k0 + 16 + hi);
        v8h b20 = *(const v8h*)(brow2 + k0 + hi);
        v8h b21 = *(const v8h*)(brow2 + k0 + 16 + hi);
        v8h b30 = *(const v8h*)(brow3 + k0 + hi);
        v8h b31 = *(const v8h*)(brow3 + k0 + 16 + hi);
        v16h bv0 = __builtin_shufflevector(b00, b01, 0,1,2,3,4,5,6,7,8,9,10,11,12,13,14,15);
        v16h bv1 = __builtin_shufflevector(b10, b11, 0,1,2,3,4,5,6,7,8,9,10,11,12,13,14,15);
        v16h bv2 = __builtin_shufflevector(b20, b21, 0,1,2,3,4,5,6,7,8,9,10,11,12,13,14,15);
        v16h bv3 = __builtin_shufflevector(b30, b31, 0,1,2,3,4,5,6,7,8,9,10,11,12,13,14,15);

        SCHED_FENCE();   // nothing crosses: loads above, WMMAs below

        acc0 = __builtin_amdgcn_wmma_f32_16x16x32_f16(
            false, av, false, bv0, (short)0, acc0, false, false);
        acc1 = __builtin_amdgcn_wmma_f32_16x16x32_f16(
            false, av, false, bv1, (short)0, acc1, false, false);
        acc2 = __builtin_amdgcn_wmma_f32_16x16x32_f16(
            false, av, false, bv2, (short)0, acc2, false, false);
        acc3 = __builtin_amdgcn_wmma_f32_16x16x32_f16(
            false, av, false, bv3, (short)0, acc3, false, false);
    }

    const int rbase = (ms << 7) + (wave << 4) + hi;  // C layout: VGPR r -> row rbase+r
    v8f accs[4] = {acc0, acc1, acc2, acc3};
#pragma unroll
    for (int nt = 0; nt < 4; ++nt) {
        const int col = n_base + (nt << 4) + l15;
        const float bb = bias[col];
#pragma unroll
        for (int r = 0; r < 8; ++r) {
            int row = rbase + r;
            if (row < M) C[(size_t)row * N + col] = accs[nt][r] + bb;
        }
    }
}

// ---------------------------------------------------------------------------
// Softmax bookkeeping init
// ---------------------------------------------------------------------------
__global__ void k_init_nodes(float* __restrict__ maxv, float* __restrict__ sumv, int n) {
    int i = blockIdx.x * blockDim.x + threadIdx.x;
    if (i < n) { maxv[i] = -__builtin_huge_valf(); sumv[i] = 0.0f; }
}

// ---------------------------------------------------------------------------
// Layer-1 edge score: one wave per edge (d = 128, lane owns a float4 slice)
// qkvs row layout: [q:0..127 | k:128..255 | v:256..383 | out/skip:384..511]
// ---------------------------------------------------------------------------
__global__ void k_edge_score_max128(const long long* __restrict__ src,
                                    const long long* __restrict__ dst,
                                    const float* __restrict__ qkvs,
                                    float* __restrict__ esc,
                                    float* __restrict__ maxv, int E) {
    int e = blockIdx.x * (blockDim.x >> 5) + (threadIdx.x >> 5);
    if (e >= E) return;
    int lane = threadIdx.x & 31;
    int s = (int)src[e], d = (int)dst[e];
    const float4 qv = *(const float4*)(qkvs + (size_t)d * 512 + 0   + lane * 4);
    const float4 kv = *(const float4*)(qkvs + (size_t)s * 512 + 128 + lane * 4);
    // warm L2 for the aggregation pass (global_prefetch_b8)
    __builtin_prefetch(qkvs + (size_t)s * 512 + 256 + lane * 4, 0, 0);
    float p = qv.x * kv.x + qv.y * kv.y + qv.z * kv.z + qv.w * kv.w;
#pragma unroll
    for (int off = 16; off > 0; off >>= 1) p += __shfl_xor(p, off, 32);
    p *= 0.08838834764831845f; // 1/sqrt(128)
    if (lane == 0) { esc[e] = p; atomicMaxF(maxv + d, p); }
}

// ---------------------------------------------------------------------------
// exp(score - max[dst]) + segment sum (shared by both layers)
// ---------------------------------------------------------------------------
__global__ void k_edge_exp_sum(const long long* __restrict__ dst,
                               float* __restrict__ esc,
                               const float* __restrict__ maxv,
                               float* __restrict__ sumv, int E) {
    int e = blockIdx.x * blockDim.x + threadIdx.x;
    if (e >= E) return;
    int d = (int)dst[e];
    float ex = __expf(esc[e] - maxv[d]);
    esc[e] = ex;
    atomicAdd(sumv + d, ex);
}

// ---------------------------------------------------------------------------
// Layer-1 aggregation: out[dst] += alpha * v[src]; accumulates on top of the
// skip term (x@Ws+bs) already resident in cols 384..511.
// ---------------------------------------------------------------------------
__global__ void k_edge_agg128(const long long* __restrict__ src,
                              const long long* __restrict__ dst,
                              const float* __restrict__ esc,
                              const float* __restrict__ sumv,
                              float* __restrict__ qkvs, int E) {
    int e = blockIdx.x * (blockDim.x >> 5) + (threadIdx.x >> 5);
    if (e >= E) return;
    int lane = threadIdx.x & 31;
    int s = (int)src[e], d = (int)dst[e];
    float alpha = esc[e] / sumv[d];
    const float4 vv = *(const float4*)(qkvs + (size_t)s * 512 + 256 + lane * 4);
    float* o = qkvs + (size_t)d * 512 + 384 + lane * 4;
    atomicAdd(o + 0, alpha * vv.x);
    atomicAdd(o + 1, alpha * vv.y);
    atomicAdd(o + 2, alpha * vv.z);
    atomicAdd(o + 3, alpha * vv.w);
}

// ---------------------------------------------------------------------------
// h = relu(layer1 out), converted to f16 for the layer-2 WMMA GEMM
// ---------------------------------------------------------------------------
__global__ void k_relu_to_f16(const float* __restrict__ qkvs,
                              half_t* __restrict__ hh, int total) {
    int t = blockIdx.x * blockDim.x + threadIdx.x;
    if (t >= total) return;
    int i = t >> 7, j = t & 127;
    float v = qkvs[(size_t)i * 512 + 384 + j];
    hh[t] = (half_t)fmaxf(v, 0.0f);
}

// ---------------------------------------------------------------------------
// Layer-2 edge score: d = 16, one thread per edge
// qkvs2 row layout: [q:0..15 | k:16..31 | v:32..47 | skip:48..63]
// ---------------------------------------------------------------------------
__global__ void k_edge_score_max16(const long long* __restrict__ src,
                                   const long long* __restrict__ dst,
                                   const float* __restrict__ qkvs2,
                                   float* __restrict__ esc,
                                   float* __restrict__ maxv, int E) {
    int e = blockIdx.x * blockDim.x + threadIdx.x;
    if (e >= E) return;
    int s = (int)src[e], d = (int)dst[e];
    const float* q = qkvs2 + (size_t)d * 64;
    const float* k = qkvs2 + (size_t)s * 64 + 16;
    __builtin_prefetch(qkvs2 + (size_t)s * 64 + 32, 0, 0);
    float p = 0.0f;
#pragma unroll
    for (int j = 0; j < 16; ++j) p += q[j] * k[j];
    p *= 0.25f; // 1/sqrt(16)
    esc[e] = p;
    atomicMaxF(maxv + d, p);
}

// ---------------------------------------------------------------------------
// Seed d_out with the layer-2 skip term
// ---------------------------------------------------------------------------
__global__ void k_copy_skip(const float* __restrict__ qkvs2,
                            float* __restrict__ out, int total) {
    int t = blockIdx.x * blockDim.x + threadIdx.x;
    if (t >= total) return;
    int i = t >> 4, j = t & 15;
    out[t] = qkvs2[(size_t)i * 64 + 48 + j];
}

// ---------------------------------------------------------------------------
// Layer-2 aggregation directly into d_out: one thread per (edge, dim)
// ---------------------------------------------------------------------------
__global__ void k_edge_agg16(const long long* __restrict__ src,
                             const long long* __restrict__ dst,
                             const float* __restrict__ esc,
                             const float* __restrict__ sumv,
                             const float* __restrict__ qkvs2,
                             float* __restrict__ out, int total) {
    int t = blockIdx.x * blockDim.x + threadIdx.x;
    if (t >= total) return;
    int e = t >> 4, j = t & 15;
    int s = (int)src[e], d = (int)dst[e];
    float alpha = esc[e] / sumv[d];
    atomicAdd(out + (size_t)d * 16 + j, alpha * qkvs2[(size_t)s * 64 + 32 + j]);
}

// ---------------------------------------------------------------------------
// Host-side orchestration
// ---------------------------------------------------------------------------
extern "C" void kernel_launch(void* const* d_in, const int* in_sizes, int n_in,
                              void* d_out, int out_size, void* d_ws, size_t ws_size,
                              hipStream_t stream) {
    (void)in_sizes; (void)n_in; (void)out_size; (void)ws_size;

    const float* x = (const float*)d_in[0];
    const long long* ei = (const long long*)d_in[1];   // int64 edge index
    const long long* src = ei;
    const long long* dst = ei + N_EDGES;

    const float* Wq1 = (const float*)d_in[2];  const float* bq1 = (const float*)d_in[3];
    const float* Wk1 = (const float*)d_in[4];  const float* bk1 = (const float*)d_in[5];
    const float* Wv1 = (const float*)d_in[6];  const float* bv1 = (const float*)d_in[7];
    const float* Ws1 = (const float*)d_in[8];  const float* bs1 = (const float*)d_in[9];
    const float* Wq2 = (const float*)d_in[10]; const float* bq2 = (const float*)d_in[11];
    const float* Wk2 = (const float*)d_in[12]; const float* bk2 = (const float*)d_in[13];
    const float* Wv2 = (const float*)d_in[14]; const float* bv2 = (const float*)d_in[15];
    const float* Ws2 = (const float*)d_in[16]; const float* bs2 = (const float*)d_in[17];

    float* out = (float*)d_out;

    // ---- workspace carve-up (all chunks 16B aligned) ----
    char* ws = (char*)d_ws;
    half_t* xh    = (half_t*)ws; ws += (size_t)N_NODES * D_IN  * sizeof(half_t); // 51.2 MB
    half_t* hh    = (half_t*)ws; ws += (size_t)N_NODES * D_H1  * sizeof(half_t); // 25.6 MB
    half_t* W1t   = (half_t*)ws; ws += (size_t)(4 * D_H1) * D_IN * sizeof(half_t);
    float*  b1    = (float*) ws; ws += (size_t)(4 * D_H1) * sizeof(float);
    half_t* W2t   = (half_t*)ws; ws += (size_t)(4 * D_H2) * D_H1 * sizeof(half_t);
    float*  b2    = (float*) ws; ws += (size_t)(4 * D_H2) * sizeof(float);
    float*  qkvs1 = (float*) ws; ws += (size_t)N_NODES * (4 * D_H1) * sizeof(float); // 204.8 MB
    float*  qkvs2 = (float*) ws; ws += (size_t)N_NODES * (4 * D_H2) * sizeof(float); // 25.6 MB
    float*  maxv  = (float*) ws; ws += (size_t)N_NODES * sizeof(float);
    float*  sumv  = (float*) ws; ws += (size_t)N_NODES * sizeof(float);
    float*  esc   = (float*) ws; ws += (size_t)N_EDGES * sizeof(float);             // 6.4 MB

    const int TB = 256;

    // ---- prep: f16 conversions + packed/transposed weights ----
    k_f32_to_f16<<<(N_NODES * D_IN + TB - 1) / TB, TB, 0, stream>>>(x, xh, N_NODES * D_IN);
    k_pack_w<<<(4 * D_H1 * D_IN + TB - 1) / TB, TB, 0, stream>>>(
        Wq1, Wk1, Wv1, Ws1, bq1, bk1, bv1, bs1, W1t, b1, D_IN, D_H1);
    k_pack_w<<<(4 * D_H2 * D_H1 + TB - 1) / TB, TB, 0, stream>>>(
        Wq2, Wk2, Wv2, Ws2, bq2, bk2, bv2, bs2, W2t, b2, D_H1, D_H2);

    // ---- layer 1 GEMM: [100000 x 256] x [256 x 512] -> qkvs1 ----
    {
        int m_strips = (N_NODES + 127) / 128;          // 782
        int n_strips = (4 * D_H1) / 64;                // 8
        k_wmma_gemm<<<m_strips * n_strips, TB, 0, stream>>>(
            xh, W1t, b1, qkvs1, N_NODES, D_IN, 4 * D_H1);
    }

    // ---- layer 1 edge phase ----
    k_init_nodes<<<(N_NODES + TB - 1) / TB, TB, 0, stream>>>(maxv, sumv, N_NODES);
    k_edge_score_max128<<<(N_EDGES + 7) / 8, TB, 0, stream>>>(src, dst, qkvs1, esc, maxv, N_EDGES);
    k_edge_exp_sum<<<(N_EDGES + TB - 1) / TB, TB, 0, stream>>>(dst, esc, maxv, sumv, N_EDGES);
    k_edge_agg128<<<(N_EDGES + 7) / 8, TB, 0, stream>>>(src, dst, esc, sumv, qkvs1, N_EDGES);

    // ---- relu + f16 convert -> layer-2 GEMM input ----
    k_relu_to_f16<<<(N_NODES * D_H1 + TB - 1) / TB, TB, 0, stream>>>(qkvs1, hh, N_NODES * D_H1);

    // ---- layer 2 GEMM: [100000 x 128] x [128 x 64] -> qkvs2 ----
    {
        int m_strips = (N_NODES + 127) / 128;          // 782
        int n_strips = (4 * D_H2) / 64;                // 1
        k_wmma_gemm<<<m_strips * n_strips, TB, 0, stream>>>(
            hh, W2t, b2, qkvs2, N_NODES, D_H1, 4 * D_H2);
    }

    // ---- layer 2 edge phase, accumulating straight into d_out ----
    k_init_nodes<<<(N_NODES + TB - 1) / TB, TB, 0, stream>>>(maxv, sumv, N_NODES);
    k_edge_score_max16<<<(N_EDGES + TB - 1) / TB, TB, 0, stream>>>(src, dst, qkvs2, esc, maxv, N_EDGES);
    k_edge_exp_sum<<<(N_EDGES + TB - 1) / TB, TB, 0, stream>>>(dst, esc, maxv, sumv, N_EDGES);
    k_copy_skip<<<(N_NODES * D_H2 + TB - 1) / TB, TB, 0, stream>>>(qkvs2, out, N_NODES * D_H2);
    k_edge_agg16<<<(N_EDGES * 16 + TB - 1) / TB, TB, 0, stream>>>(
        src, dst, esc, sumv, qkvs2, out, N_EDGES * 16);
}
